// Classifier_14602888806638
// MI455X (gfx1250) — compile-verified, matrix operand order checked by reading, not run on hardware
//
#include <hip/hip_runtime.h>
#include <hip/hip_bf16.h>

#define HIDDEN 768
#define VOCAB  32000
#define HS1    128
#define KC_CNT (HIDDEN / 32)                    // 24 K-chunks of 32
#define NT_CNT (HS1 / 16)                       // 8 N-tiles of 16
#define B_HALF_TOTAL (KC_CNT * NT_CNT * 32 * 16)  // 98304 f16 elements
#define B_BYTES (B_HALF_TOTAL * 2)              // 196608 bytes (fits 320KB LDS)

typedef _Float16 h16;
typedef __attribute__((ext_vector_type(16))) _Float16 v16h;
typedef __attribute__((ext_vector_type(8)))  float    v8f;
typedef __attribute__((ext_vector_type(4)))  float    v4f;

// ---------------------------------------------------------------------------
// Prep: convert W1[VOCAB:] (768x128 fp32, row-major) to f16, laid out in the
// exact per-lane B-fragment order for V_WMMA_F32_16X16X32_F16 so the main
// kernel reads each fragment as one contiguous 32B LDS vector per lane.
//
// Fragment slot j (0..15) of lane l for K-chunk kc, N-tile nt maps to:
//   n = nt*16 + (l & 15)
//   k = kc*32 + (j>>3)*16 + (l>>4)*8 + (j & 7)
// ---------------------------------------------------------------------------
__global__ void prep_w1h_f16_swizzle(const float* __restrict__ W1,
                                     h16* __restrict__ Bws) {
    int idx = blockIdx.x * blockDim.x + threadIdx.x;
    if (idx >= B_HALF_TOTAL) return;
    int j    = idx & 15;
    int lane = (idx >> 4) & 31;
    int nt   = (idx >> 9) & 7;
    int kc   = idx >> 12;
    int n = nt * 16 + (lane & 15);
    int k = kc * 32 + ((j >> 3) << 4) + ((lane >> 4) << 3) + (j & 7);
    Bws[idx] = (h16)W1[(size_t)(VOCAB + k) * HS1 + n];
}

// ---------------------------------------------------------------------------
// Main fused kernel: 8 waves per workgroup, each wave computes one 16-token
// tile of hs0 @ W1h via v_wmma_f32_16x16x32_f16 (fp32 accumulate), then fuses
// the one-hot gather W1[tk], +b1, ReLU, dot(W2), +b2 in the epilogue.
// B staging uses CDNA5 async global->LDS DMA (no VGPR round-trip).
// ---------------------------------------------------------------------------
__global__ __launch_bounds__(256)
void classifier_wmma(const long long* __restrict__ tk,
                     const float* __restrict__ hs0,
                     const float* __restrict__ W1,
                     const float* __restrict__ b1,
                     const float* __restrict__ W2,
                     const float* __restrict__ b2,
                     const h16* __restrict__ Bws,
                     float* __restrict__ out,
                     int n_tokens) {
    extern __shared__ char smem_raw[];

    // ---- async stage: 196KB pre-swizzled f16 W1h, global -> LDS ----------
    // GLOBAL_LOAD_ASYNC_TO_LDS_B128 (GVS mode): SGPR base = Bws,
    // VGPR %1 = byte offset into Bws, VGPR %0 = LDS byte address.
    // Low 32 bits of a generic shared-space pointer are the LDS byte address
    // (flat aperture rule).
    {
        unsigned lds_base = (unsigned)(uintptr_t)smem_raw;
        for (unsigned i = threadIdx.x; i < B_BYTES / 16; i += 256) {
            unsigned goff  = i * 16u;
            unsigned laddr = lds_base + goff;
            asm volatile(
                "global_load_async_to_lds_b128 %0, %1, %2"
                :
                : "v"(laddr), "v"(goff), "s"(Bws)
                : "memory");
        }
        asm volatile("s_wait_asynccnt 0" ::: "memory");
    }
    __syncthreads();

    const v16h* __restrict__ bsm = (const v16h*)smem_raw;

    const int lane = threadIdx.x & 31;
    const int wave = threadIdx.x >> 5;
    const int tile = blockIdx.x * 8 + wave;
    const int row0 = tile * 16;                 // first token of this wave's tile
    if (row0 >= n_tokens) return;

    const int nlo   = lane & 15;
    const int khalf = lane >> 4;                // 0: lanes 0-15, 1: lanes 16-31

    // A-fragment source row for this lane: token row (row0 + nlo), K offset by
    // 8 for the upper half-wave (16-bit A-matrix layout).
    const float* __restrict__ arow =
        hs0 + (size_t)(row0 + nlo) * HIDDEN + (khalf << 3);

    v8f acc[NT_CNT] = {};                       // 16x128 fp32 tile, 64 VGPRs

    for (int kc = 0; kc < KC_CNT; ++kc) {
        const float* ap = arow + kc * 32;
        if (kc + 1 < KC_CNT)
            __builtin_prefetch(ap + 32, 0, 1);  // -> global_prefetch_b8

        v4f a0 = *(const v4f*)(ap);             // K = base+0..3
        v4f a1 = *(const v4f*)(ap + 4);         // K = base+4..7
        v4f a2 = *(const v4f*)(ap + 16);        // K = base+16..19
        v4f a3 = *(const v4f*)(ap + 20);        // K = base+20..23

        v16h a;
        #pragma unroll
        for (int i = 0; i < 4; ++i) {
            a[i]      = (h16)a0[i];
            a[4 + i]  = (h16)a1[i];
            a[8 + i]  = (h16)a2[i];
            a[12 + i] = (h16)a3[i];
        }

        #pragma unroll
        for (int nt = 0; nt < NT_CNT; ++nt) {
            v16h bfrag = bsm[(kc * NT_CNT + nt) * 32 + lane];
            acc[nt] = __builtin_amdgcn_wmma_f32_16x16x32_f16(
                false, a, false, bfrag, (short)0, acc[nt], false, false);
        }
    }

    // ---------------- fused epilogue ----------------
    // C/D layout: acc[nt] VGPR r at this lane = C[m = r + 8*khalf][n = nt*16+nlo]
    float b1v[NT_CNT], w2v[NT_CNT];
    #pragma unroll
    for (int nt = 0; nt < NT_CNT; ++nt) {
        b1v[nt] = b1[nt * 16 + nlo];
        w2v[nt] = W2[nt * 16 + nlo];
    }
    const float b2v = b2[0];
    const int mbase = row0 + (khalf << 3);

    #pragma unroll
    for (int r = 0; r < 8; ++r) {
        const int trow = mbase + r;
        float p = 0.0f;
        if (trow < n_tokens) {
            const long long tok = tk[trow];
            const float* __restrict__ g = W1 + (size_t)tok * HS1 + nlo;
            #pragma unroll
            for (int nt = 0; nt < NT_CNT; ++nt) {
                float hv = acc[nt][r] + g[nt * 16] + b1v[nt];
                hv = hv > 0.0f ? hv : 0.0f;
                p += hv * w2v[nt];
            }
        }
        // reduce over the 16 lanes of each half-wave (wave32)
        p += __shfl_xor(p, 1, 32);
        p += __shfl_xor(p, 2, 32);
        p += __shfl_xor(p, 4, 32);
        p += __shfl_xor(p, 8, 32);
        if (nlo == 0 && trow < n_tokens)
            out[trow] = p + b2v;
    }
}

extern "C" void kernel_launch(void* const* d_in, const int* in_sizes, int n_in,
                              void* d_out, int out_size, void* d_ws, size_t ws_size,
                              hipStream_t stream) {
    (void)n_in; (void)out_size; (void)ws_size;

    const long long* tk  = (const long long*)d_in[0];  // int64 tokens [B,S]
    const float*     hs0 = (const float*)d_in[1];      // [B,S,768]
    const float*     W1  = (const float*)d_in[2];      // [32768,128]
    const float*     b1  = (const float*)d_in[3];      // [128]
    const float*     W2  = (const float*)d_in[4];      // [128,1]
    const float*     b2  = (const float*)d_in[5];      // [1]
    float* out = (float*)d_out;

    const int n_tokens = in_sizes[0];                  // B*S = 65536
    h16* Bws = (h16*)d_ws;                             // 196608 bytes scratch

    // 1) one-time f16 swizzle of W1h into workspace
    prep_w1h_f16_swizzle<<<(B_HALF_TOTAL + 255) / 256, 256, 0, stream>>>(W1, Bws);

    // 2) fused gather + GEMM(WMMA f16) + ReLU + dot reduction
    const int tiles  = (n_tokens + 15) / 16;
    const int blocks = (tiles + 7) / 8;
    classifier_wmma<<<blocks, 256, B_BYTES, stream>>>(
        tk, hs0, W1, b1, W2, b2, Bws, out, n_tokens);
}
